// MessagePassingModel_40707700031636
// MI455X (gfx1250) — matrix operand order repaired and checked by reading, not compiled, not run on hardware
//
#include <hip/hip_runtime.h>
#include <hip/hip_bf16.h>
#include <math.h>

// ---------------------------------------------------------------------------
// Equivariant MPNN (l<=2, 9 channels x 32 features), 3 iterations + head.
// X, Y node-state buffers live in d_ws (2 x N*9*32 floats = 23 MB).
// Node dense layers use v_wmma_f32_16x16x32_f16 (f16 in, f32 accumulate).
// ---------------------------------------------------------------------------

typedef __attribute__((ext_vector_type(16))) _Float16 v16h;
typedef __attribute__((ext_vector_type(8)))  float    v8f;

#define FDIM 32
#define NCH  9

// silu via hardware v_rcp_f32 (avoids IEEE div fix-up sequence)
__device__ __forceinline__ float siluf(float x)
{
    return x * __builtin_amdgcn_rcpf(1.0f + __expf(-x));
}

// ---- Gaunt coefficient magnitudes (analytic, match reference quadrature) ----
#define G_C0  0.28209479177387814f   // 1/(2 sqrt(pi))
#define G_CA  0.21850968611841584f   // c2a/5
#define G_B1  0.12615662610100802f   // 0.4*c2b   (sign applied in code)
#define G_B2  0.25231325220201604f   // 0.8*c2b
#define G_T1  0.18022375157286857f   // (4/7)*c2b
#define G_T2  0.09011187578643429f   // (2/7)*c2b
#define G_T3  0.15607834722743988f   // (2/7)*c2c

// ---------------------------------------------------------------------------
// Init: X[:,0,:] = embed[Z], other channels 0; Y = X.
// ---------------------------------------------------------------------------
__global__ void mp_init_kernel(const int* __restrict__ Z,
                               const float* __restrict__ embed,
                               float* __restrict__ X, float* __restrict__ Y, int N)
{
    int idx = blockIdx.x * blockDim.x + threadIdx.x;
    int total = N * NCH * FDIM;
    if (idx >= total) return;
    int f = idx & (FDIM - 1);
    int c = (idx >> 5) % NCH;
    int n = idx / (NCH * FDIM);
    float v = (c == 0) ? embed[Z[n] * FDIM + f] : 0.0f;
    X[idx] = v;
    Y[idx] = v;
}

// ---------------------------------------------------------------------------
// Edge kernel: one wave32 per edge, lane = feature.
// Wb slice for this iteration (3*16*32 f32 = 6 KB) is staged in LDS per block.
//   bf[b,f] = sph[b] * (sum_k rad[k] * Wb[i][DEG[b]][k][f])  (DEG-collapsed)
//   msg[c,f] = sum_{a,b} CG[a,b,c] x[src,a,f] bf[b,f]  -> atomicAdd into Y[dst]
// ---------------------------------------------------------------------------
__global__ __launch_bounds__(256)
void mp_edge_kernel(const float* __restrict__ pos,
                    const int* __restrict__ dst_idx,
                    const int* __restrict__ src_idx,
                    const float* __restrict__ Wb,
                    const float* __restrict__ X,
                    float* __restrict__ Y,
                    int iter, int E)
{
    __shared__ float swb[3 * 16 * FDIM];          // 6 KB of 320 KB WGP LDS
    {
        const float* wbg = Wb + (size_t)iter * 3 * 16 * FDIM;
        for (int i = threadIdx.x; i < 3 * 16 * FDIM; i += 256)
            swb[i] = wbg[i];
    }
    __syncthreads();                              // before ANY divergent return

    const int e = blockIdx.x * 8 + (threadIdx.x >> 5);
    if (e >= E) return;                           // wave-uniform
    const int lane = threadIdx.x & 31;

    const int s = src_idx[e];
    const int d = dst_idx[e];

    const float dx = pos[s * 3 + 0] - pos[d * 3 + 0];
    const float dy = pos[s * 3 + 1] - pos[d * 3 + 1];
    const float dz = pos[s * 3 + 2] - pos[d * 3 + 2];
    const float r  = sqrtf(dx * dx + dy * dy + dz * dz + 1e-12f);
    const float ir = 1.0f / r;
    const float ux = dx * ir, uy = dy * ir, uz = dz * ir;

    // cutoff envelope (uniform over wave -> uniform early exit)
    const float t = r * 0.25f;
    float fc = 0.0f;
    if (t < 1.0f) {
        const float den = 1.0f - t * t;
        fc = __expf(-t * t / den);
    }
    if (fc == 0.0f) return;                       // msg == 0 entirely

    // real spherical harmonics l<=2 (reference ordering)
    const float c0  = 0.28209479177387814f;
    const float c1  = 0.4886025119029199f;
    const float c2a = 1.0925484305920792f;
    const float c2b = 0.31539156525252005f;
    const float c2c = 0.5462742152960396f;
    float sph[9];
    sph[0] = c0;
    sph[1] = c1 * uy;
    sph[2] = c1 * uz;
    sph[3] = c1 * ux;
    sph[4] = c2a * ux * uy;
    sph[5] = c2a * uy * uz;
    sph[6] = c2b * (3.0f * uz * uz - 1.0f);
    sph[7] = c2a * ux * uz;
    sph[8] = c2c * (ux * ux - uy * uy);

    // Bernstein radial basis * envelope
    const float BIN[16] = {1.f, 15.f, 105.f, 455.f, 1365.f, 3003.f, 5005.f, 6435.f,
                           6435.f, 5005.f, 3003.f, 1365.f, 455.f, 105.f, 15.f, 1.f};
    const float u1 = 1.0f / (1.0f + r);
    const float v1 = 1.0f - u1;
    float pu[16], pv[16];
    pu[0] = 1.0f; pv[0] = 1.0f;
    #pragma unroll
    for (int k = 1; k < 16; ++k) { pu[k] = pu[k - 1] * u1; pv[k] = pv[k - 1] * v1; }
    float rad[16];
    #pragma unroll
    for (int k = 0; k < 16; ++k) rad[k] = BIN[k] * pu[k] * pv[15 - k] * fc;

    // rw[d][lane] = sum_k rad[k] * Wb[iter][d][k][lane]  (from LDS)
    float rw0 = 0.0f, rw1 = 0.0f, rw2 = 0.0f;
    #pragma unroll
    for (int k = 0; k < 16; ++k) {
        rw0 = fmaf(rad[k], swb[(0 * 16 + k) * FDIM + lane], rw0);
        rw1 = fmaf(rad[k], swb[(1 * 16 + k) * FDIM + lane], rw1);
        rw2 = fmaf(rad[k], swb[(2 * 16 + k) * FDIM + lane], rw2);
    }
    const float b0 = sph[0] * rw0;
    const float b1 = sph[1] * rw1, b2 = sph[2] * rw1, b3 = sph[3] * rw1;
    const float b4 = sph[4] * rw2, b5 = sph[5] * rw2, b6 = sph[6] * rw2;
    const float b7 = sph[7] * rw2, b8 = sph[8] * rw2;

    const float* xs = X + ((size_t)s * NCH) * FDIM + lane;
    const float x0 = xs[0 * FDIM], x1 = xs[1 * FDIM], x2 = xs[2 * FDIM];
    const float x3 = xs[3 * FDIM], x4 = xs[4 * FDIM], x5 = xs[5 * FDIM];
    const float x6 = xs[6 * FDIM], x7 = xs[7 * FDIM], x8 = xs[8 * FDIM];

    // msg[c] = sum_{a,b} CG[a,b,c] x_a b_b  (fully unrolled symmetric Gaunt tensor)
    float m[9];
    m[0] = G_C0 * (x0*b0 + x1*b1 + x2*b2 + x3*b3 + x4*b4 + x5*b5 + x6*b6 + x7*b7 + x8*b8);
    m[1] = G_C0 * (x0*b1 + x1*b0)
         + G_CA * (x3*b4 + x4*b3 + x2*b5 + x5*b2)
         - G_B1 * (x1*b6 + x6*b1)
         - G_CA * (x1*b8 + x8*b1);
    m[2] = G_C0 * (x0*b2 + x2*b0)
         + G_CA * (x1*b5 + x5*b1 + x3*b7 + x7*b3)
         + G_B2 * (x2*b6 + x6*b2);
    m[3] = G_C0 * (x0*b3 + x3*b0)
         + G_CA * (x1*b4 + x4*b1 + x2*b7 + x7*b2)
         - G_B1 * (x3*b6 + x6*b3)
         + G_CA * (x3*b8 + x8*b3);
    m[4] = G_C0 * (x0*b4 + x4*b0)
         + G_CA * (x1*b3 + x3*b1)
         - G_T1 * (x4*b6 + x6*b4)
         + G_T3 * (x5*b7 + x7*b5);
    m[5] = G_C0 * (x0*b5 + x5*b0)
         + G_CA * (x1*b2 + x2*b1)
         + G_T2 * (x5*b6 + x6*b5)
         - G_T3 * (x5*b8 + x8*b5)
         + G_T3 * (x4*b7 + x7*b4);
    m[6] = G_C0 * (x0*b6 + x6*b0)
         - G_B1 * (x1*b1 + x3*b3)
         + G_B2 * (x2*b2)
         - G_T1 * (x4*b4)
         + G_T2 * (x5*b5 + x7*b7)
         + G_T1 * (x6*b6)
         - G_T1 * (x8*b8);
    m[7] = G_C0 * (x0*b7 + x7*b0)
         + G_CA * (x2*b3 + x3*b2)
         + G_T2 * (x7*b6 + x6*b7)
         + G_T3 * (x7*b8 + x8*b7)
         + G_T3 * (x4*b5 + x5*b4);
    m[8] = G_C0 * (x0*b8 + x8*b0)
         - G_CA * (x1*b1)
         + G_CA * (x3*b3)
         - G_T3 * (x5*b5)
         + G_T3 * (x7*b7)
         - G_T1 * (x6*b8 + x8*b6);

    float* yd = Y + ((size_t)d * NCH) * FDIM + lane;
    #pragma unroll
    for (int c = 0; c < NCH; ++c) atomicAdd(&yd[c * FDIM], m[c]);
}

// ---------------------------------------------------------------------------
// Node kernel: one wave per 16-node tile. Loop over degrees; the four f16
// B fragments (W1/W2 x 2 N-tiles) are loaded once per degree and kept in
// VGPRs across all channels of that degree. Per channel:
//   h = silu(Y_tile(16x32) @ W1[d] + b1*(c==0)); out = h @ W2[d] + b2*(c==0)
//   Xnew = X + out; write to X and Y.
// GEMMs via v_wmma_f32_16x16x32_f16, LDS bounce for the D->A relayout.
// ---------------------------------------------------------------------------
__global__ __launch_bounds__(256)
void mp_node_kernel(const float* __restrict__ W1, const float* __restrict__ B1,
                    const float* __restrict__ W2, const float* __restrict__ B2,
                    float* __restrict__ X, float* __restrict__ Y,
                    int iter, int nTiles, int N)
{
    __shared__ float hbuf[8][16 * FDIM];
    const int wid  = threadIdx.x >> 5;
    const int lane = threadIdx.x & 31;
    const int tile = blockIdx.x * 8 + wid;
    if (tile >= nTiles) return;               // wave-uniform, EXEC stays all-1s

    const int node0 = tile * 16;
    const int row   = lane & 15;
    const int hi    = lane >> 4;              // K half this lane holds (A layout)
    const int kb    = 8 * hi;
    int nodeA = node0 + row; if (nodeA >= N) nodeA = N - 1;   // clamp loads
    float* hs = hbuf[wid];

    // biases (only applied at channel 0); col per N-tile is lane-constant
    float bias1[2], bias2[2];
    #pragma unroll
    for (int t = 0; t < 2; ++t) {
        const int col = (lane & 15) + 16 * t;
        bias1[t] = B1[iter * FDIM + col];
        bias2[t] = B2[iter * FDIM + col];
    }

    const int cbeg[3] = {0, 1, 4};
    const int cend[3] = {1, 4, 9};

    for (int dg = 0; dg < 3; ++dg) {
        const float* w1 = W1 + ((size_t)(iter * 3 + dg) * FDIM) * FDIM;
        const float* w2 = W2 + ((size_t)(iter * 3 + dg) * FDIM) * FDIM;

        // ---- B fragments for this degree (held in VGPRs across channels) ----
        v16h bw1[2], bw2[2];
        #pragma unroll
        for (int t = 0; t < 2; ++t) {
            const int col = (lane & 15) + 16 * t;
            #pragma unroll
            for (int j = 0; j < 16; ++j) {
                bw1[t][j] = (_Float16)w1[(kb + j + (j >= 8 ? 8 : 0)) * FDIM + col];
                bw2[t][j] = (_Float16)w2[(kb + j + (j >= 8 ? 8 : 0)) * FDIM + col];
            }
        }

        for (int c = cbeg[dg]; c < cend[dg]; ++c) {
            // ---- A <- Y tile (f32 -> f16), ISA 16-bit A layout ----
            const float* ysrc = Y + ((size_t)nodeA * NCH + c) * FDIM;
            v16h a;
            #pragma unroll
            for (int j = 0; j < 8; ++j) {
                a[j]     = (_Float16)ysrc[kb + j];
                a[j + 8] = (_Float16)ysrc[16 + kb + j];
            }

            // ---- layer 1: two 16-wide N tiles ----
            #pragma unroll
            for (int t = 0; t < 2; ++t) {
                const int col = (lane & 15) + 16 * t;
                v8f acc = {};
                acc = __builtin_amdgcn_wmma_f32_16x16x32_f16(
                          false, a, false, bw1[t], (short)0, acc, false, false);
                const float bias = (c == 0) ? bias1[t] : 0.0f;
                #pragma unroll
                for (int j = 0; j < 8; ++j) {
                    const int mrow = j + 8 * hi;
                    hs[mrow * FDIM + col] = siluf(acc[j] + bias);
                }
            }
            asm volatile("s_wait_dscnt 0" ::: "memory");

            // ---- A2 <- LDS h tile (transpose bounce) ----
            const float* hrow = hs + row * FDIM;
            v16h a2;
            #pragma unroll
            for (int j = 0; j < 8; ++j) {
                a2[j]     = (_Float16)hrow[kb + j];
                a2[j + 8] = (_Float16)hrow[16 + kb + j];
            }

            // ---- layer 2 + residual, write X and Y ----
            #pragma unroll
            for (int t = 0; t < 2; ++t) {
                const int col = (lane & 15) + 16 * t;
                v8f acc = {};
                acc = __builtin_amdgcn_wmma_f32_16x16x32_f16(
                          false, a2, false, bw2[t], (short)0, acc, false, false);
                const float bias = (c == 0) ? bias2[t] : 0.0f;
                #pragma unroll
                for (int j = 0; j < 8; ++j) {
                    const int mrow = j + 8 * hi;
                    const int nidx = node0 + mrow;
                    if (nidx < N) {
                        const size_t o = ((size_t)nidx * NCH + c) * FDIM + col;
                        const float xn = X[o] + acc[j] + bias;
                        X[o] = xn;
                        Y[o] = xn;
                    }
                }
            }
        }
    }
}

// ---------------------------------------------------------------------------
// Head: mono = x[:,0,:] @ Wt00 @ Wmono + eb[Z];  dip = clip(silu(x[:,1:4,:]@Wt11)) + pos
// Output layout: [mono (N*4) | dip (N*3*4)]
// ---------------------------------------------------------------------------
__global__ void mp_head_kernel(const int* __restrict__ Z,
                               const float* __restrict__ pos,
                               const float* __restrict__ X,
                               const float* __restrict__ Wt00,
                               const float* __restrict__ Wt11,
                               const float* __restrict__ Wmono,
                               const float* __restrict__ eb,
                               float* __restrict__ out, int N)
{
    int n = blockIdx.x * blockDim.x + threadIdx.x;
    if (n >= N) return;

    const float* x0 = X + (size_t)n * NCH * FDIM;
    float p[4] = {0.f, 0.f, 0.f, 0.f};
    for (int f = 0; f < FDIM; ++f) {
        const float xv = x0[f];
        #pragma unroll
        for (int j = 0; j < 4; ++j) p[j] = fmaf(xv, Wt00[f * 4 + j], p[j]);
    }
    const float ebv = eb[Z[n]];
    #pragma unroll
    for (int mcol = 0; mcol < 4; ++mcol) {
        float mono = ebv;
        #pragma unroll
        for (int j = 0; j < 4; ++j) mono = fmaf(p[j], Wmono[j * 4 + mcol], mono);
        out[n * 4 + mcol] = mono;
    }

    float* dip = out + (size_t)N * 4;
    for (int cc = 0; cc < 3; ++cc) {
        const float* xc = X + ((size_t)n * NCH + 1 + cc) * FDIM;
        const float pc = pos[n * 3 + cc];
        float q[4] = {0.f, 0.f, 0.f, 0.f};
        for (int f = 0; f < FDIM; ++f) {
            const float xv = xc[f];
            #pragma unroll
            for (int g = 0; g < 4; ++g) q[g] = fmaf(xv, Wt11[f * 4 + g], q[g]);
        }
        #pragma unroll
        for (int g = 0; g < 4; ++g) {
            float v = siluf(q[g]);
            v = fminf(0.3f, fmaxf(-0.3f, v));
            dip[((size_t)n * 3 + cc) * 4 + g] = v + pc;
        }
    }
}

// ---------------------------------------------------------------------------
extern "C" void kernel_launch(void* const* d_in, const int* in_sizes, int n_in,
                              void* d_out, int out_size, void* d_ws, size_t ws_size,
                              hipStream_t stream)
{
    const int*   Z     = (const int*)  d_in[0];
    const float* pos   = (const float*)d_in[1];
    const int*   dsti  = (const int*)  d_in[2];
    const int*   srci  = (const int*)  d_in[3];
    const float* embed = (const float*)d_in[4];
    const float* Wb    = (const float*)d_in[5];
    const float* W1    = (const float*)d_in[6];
    const float* b1    = (const float*)d_in[7];
    const float* W2    = (const float*)d_in[8];
    const float* b2    = (const float*)d_in[9];
    const float* Wt00  = (const float*)d_in[10];
    const float* Wt11  = (const float*)d_in[11];
    const float* Wmono = (const float*)d_in[12];
    const float* ebias = (const float*)d_in[13];

    const int N = in_sizes[0];
    const int E = in_sizes[2];

    float* X = (float*)d_ws;
    float* Y = X + (size_t)N * NCH * FDIM;

    const int total = N * NCH * FDIM;
    mp_init_kernel<<<(total + 255) / 256, 256, 0, stream>>>(Z, embed, X, Y, N);

    const int nTiles = (N + 15) / 16;
    for (int it = 0; it < 3; ++it) {
        mp_edge_kernel<<<(E + 7) / 8, 256, 0, stream>>>(pos, dsti, srci, Wb, X, Y, it, E);
        mp_node_kernel<<<(nTiles + 7) / 8, 256, 0, stream>>>(W1, b1, W2, b2, X, Y, it, nTiles, N);
    }

    mp_head_kernel<<<(N + 255) / 256, 256, 0, stream>>>(Z, pos, X, Wt00, Wt11, Wmono,
                                                        ebias, (float*)d_out, N);
}
